// Encoder_Model_86517821210687
// MI455X (gfx1250) — matrix operand order repaired and checked by reading, not compiled, not run on hardware
//
#include <hip/hip_runtime.h>
#include <hip/hip_bf16.h>
#include <math.h>

// ---------------------------------------------------------------------------
// Types for CDNA5 WMMA (wave32): bf16 16x16x32 -> f32
// ---------------------------------------------------------------------------
typedef __bf16 bf16;
typedef bf16 v16bf __attribute__((ext_vector_type(16)));
typedef bf16 v8bf  __attribute__((ext_vector_type(8)));
typedef float v8f  __attribute__((ext_vector_type(8)));

#define BATCH 8
#define SEQ   2048
#define DMODEL 512
#define DK    64
#define DFF   2048
#define MTOT  (BATCH * SEQ)            // 16384 rows
#define NELEM ((long)MTOT * DMODEL)    // 8388608 activation elems
#define LN_EPS 1e-5f
#define SCORE_SCALE (1.0f / 32.0f)     // (512/8)/2 per reference precedence bug

static __device__ __forceinline__ v8f wmma_bf16(v16bf a, v16bf b, v8f c) {
  return __builtin_amdgcn_wmma_f32_16x16x32_bf16(false, a, false, b, (short)0, c,
                                                 false, false);
}

// A fragment: 16x32 tile of row-major bf16 src (lda elems). Lane l holds row
// l&15; K pairs per ISA layout -> two contiguous 16B loads per lane.
static __device__ __forceinline__ v16bf load_a(const bf16* src, int lda,
                                               int row0, int k0) {
  int l = threadIdx.x & 31;
  int row = l & 15, hf = l >> 4;
  const bf16* p = src + (size_t)(row0 + row) * lda + k0 + hf * 8;
  v8bf lo = *(const v8bf*)(p);
  v8bf hi = *(const v8bf*)(p + 16);
  v16bf a;
#pragma unroll
  for (int i = 0; i < 8; ++i) { a[i] = lo[i]; a[i + 8] = hi[i]; }
  return a;
}

// B fragment: 32(K)x16(N) tile, source stored TRANSPOSED row-major [N,K]
// (ldb elems); K pairs per ISA layout (same as A) -> two 16B loads per lane.
static __device__ __forceinline__ v16bf load_b(const bf16* wt, int ldb,
                                               int n0, int k0) {
  int l = threadIdx.x & 31;
  int n = l & 15, kb = (l >> 4) * 8;
  const bf16* p = wt + (size_t)(n0 + n) * ldb + k0 + kb;
  v8bf lo = *(const v8bf*)(p);
  v8bf hi = *(const v8bf*)(p + 16);
  v16bf b;
#pragma unroll
  for (int i = 0; i < 8; ++i) { b[i] = lo[i]; b[i + 8] = hi[i]; }
  return b;
}

// ---------------------------------------------------------------------------
// One-time weight prep: bf16 transposes + WoSum = sum of 8 row-blocks of Wo
// ---------------------------------------------------------------------------
__global__ __launch_bounds__(256) void prep_weights(
    const float* Wq, const float* Wk, const float* Wv, const float* Wo,
    const float* W1, const float* W2, bf16* wqT, bf16* wkT, bf16* wvT,
    bf16* woST, bf16* w1T, bf16* w2T) {
  const long NQ = 64L * 512, NOS = 512L * 64, N1 = 2048L * 512, N2 = 512L * 2048;
  long total = 3 * NQ + NOS + N1 + N2;
  for (long i = (long)blockIdx.x * blockDim.x + threadIdx.x; i < total;
       i += (long)gridDim.x * blockDim.x) {
    long j = i;
    if (j < NQ) { int n = j / 512, kk = j % 512; wqT[j] = (bf16)Wq[kk * 64 + n]; }
    else if ((j -= NQ) < NQ) { int n = j / 512, kk = j % 512; wkT[j] = (bf16)Wk[kk * 64 + n]; }
    else if ((j -= NQ) < NQ) { int n = j / 512, kk = j % 512; wvT[j] = (bf16)Wv[kk * 64 + n]; }
    else if ((j -= NQ) < NOS) {
      int e = j / 64, jj = j % 64;
      float s = 0.f;
#pragma unroll
      for (int h = 0; h < 8; ++h) s += Wo[(size_t)(h * 64 + jj) * 512 + e];
      woST[j] = (bf16)s;
    } else if ((j -= NOS) < N1) {
      int n = j / 512, kk = j % 512; w1T[j] = (bf16)W1[(size_t)kk * 2048 + n];
    } else {
      j -= N1; int n = j / 2048, kk = j % 2048; w2T[j] = (bf16)W2[(size_t)kk * 512 + n];
    }
  }
}

// ---------------------------------------------------------------------------
// Embedding + mask + interleaved sinusoidal PE -> x fp32 + bf16 copy
// ---------------------------------------------------------------------------
__global__ __launch_bounds__(256) void embed_pe(const int* ids, const float* emb,
                                                float* x, bf16* xb) {
  for (long i = (long)blockIdx.x * blockDim.x + threadIdx.x; i < NELEM;
       i += (long)gridDim.x * blockDim.x) {
    int d = (int)(i % DMODEL);
    long rs = i / DMODEL;
    int s = (int)(rs % SEQ);
    int id = ids[rs];
    float v = (id != 0) ? emb[(size_t)id * DMODEL + d] : 0.0f;
    float ang = (float)s * powf(10000.0f, -2.0f * (float)d / (float)DMODEL);
    v += (d & 1) ? cosf(ang) : sinf(ang);
    x[i] = v;
    xb[i] = (bf16)v;
  }
}

// ---------------------------------------------------------------------------
// QKV projections: one wave per 16-row tile, N=64 each for q,k,v.
// q,k stored row-major [M,64]; v stored transposed [B,64,S] for attn B-frags.
// ---------------------------------------------------------------------------
__global__ __launch_bounds__(128) void qkv_kernel(
    const bf16* xb, const bf16* wqT, const bf16* wkT, const bf16* wvT,
    const float* bq, const float* bk, const float* bv,
    bf16* q, bf16* k, bf16* vT) {
  int wave = threadIdx.x >> 5;
  int m0 = (blockIdx.x * 4 + wave) * 16;
  v8f qa[4], ka[4], va[4];
#pragma unroll
  for (int nt = 0; nt < 4; ++nt) { qa[nt] = (v8f)0.f; ka[nt] = (v8f)0.f; va[nt] = (v8f)0.f; }
  for (int kk = 0; kk < DMODEL; kk += 32) {
    v16bf a = load_a(xb, DMODEL, m0, kk);
#pragma unroll
    for (int nt = 0; nt < 4; ++nt) {
      qa[nt] = wmma_bf16(a, load_b(wqT, DMODEL, nt * 16, kk), qa[nt]);
      ka[nt] = wmma_bf16(a, load_b(wkT, DMODEL, nt * 16, kk), ka[nt]);
      va[nt] = wmma_bf16(a, load_b(wvT, DMODEL, nt * 16, kk), va[nt]);
    }
  }
  int l = threadIdx.x & 31, n = l & 15, mb = (l >> 4) * 8;
#pragma unroll
  for (int nt = 0; nt < 4; ++nt) {
#pragma unroll
    for (int r = 0; r < 8; ++r) {
      int row = m0 + mb + r, col = nt * 16 + n;
      q[(size_t)row * DK + col] = (bf16)(qa[nt][r] + bq[col]);
      k[(size_t)row * DK + col] = (bf16)(ka[nt][r] + bk[col]);
      int b = row >> 11, s = row & (SEQ - 1);
      vT[((size_t)b * DK + col) * SEQ + s] = (bf16)(va[nt][r] + bv[col]);
    }
  }
}

// ---------------------------------------------------------------------------
// Flash-style attention: one wave per (batch, 16-row q tile). Online softmax
// over 128 key tiles (2 per iter). P round-trips through 1KB LDS/wave so it
// can be reloaded in WMMA A-fragment layout. Waves stay independent (no block
// barriers): k/v stream straight from the 192MB L2.
// ---------------------------------------------------------------------------
__global__ __launch_bounds__(128) void attn_kernel(const bf16* q, const bf16* k,
                                                   const bf16* vT, bf16* head) {
  __shared__ __align__(16) bf16 pbuf[4][16 * 32];
  int wave = threadIdx.x >> 5;
  int gw = blockIdx.x * 4 + wave;        // 0..1023
  int b = gw >> 7, qt = gw & 127;
  const bf16* qp = q + (size_t)b * SEQ * DK;
  const bf16* kp = k + (size_t)b * SEQ * DK;
  const bf16* vp = vT + (size_t)b * DK * SEQ;
  int m0 = qt * 16;
  v16bf qa0 = load_a(qp, DK, m0, 0);
  v16bf qa1 = load_a(qp, DK, m0, 32);
  v8f o[4];
#pragma unroll
  for (int nc = 0; nc < 4; ++nc) o[nc] = (v8f)0.f;
  float mrow[8], lrow[8];
#pragma unroll
  for (int r = 0; r < 8; ++r) { mrow[r] = -3.0e38f; lrow[r] = 0.f; }
  bf16* P = pbuf[wave];
  int l = threadIdx.x & 31, n = l & 15, mb = (l >> 4) * 8;

  for (int it = 0; it < SEQ / 32; ++it) {
    int key0 = it * 32;
    v8f c0 = (v8f)0.f, c1 = (v8f)0.f;
    c0 = wmma_bf16(qa0, load_b(kp, DK, key0, 0), c0);
    c0 = wmma_bf16(qa1, load_b(kp, DK, key0, 32), c0);
    c1 = wmma_bf16(qa0, load_b(kp, DK, key0 + 16, 0), c1);
    c1 = wmma_bf16(qa1, load_b(kp, DK, key0 + 16, 32), c1);

    float p0[8], p1[8];
#pragma unroll
    for (int r = 0; r < 8; ++r) {
      float s0 = c0[r] * SCORE_SCALE, s1 = c1[r] * SCORE_SCALE;
      float rm = fmaxf(s0, s1);
#pragma unroll
      for (int mm = 1; mm < 16; mm <<= 1) rm = fmaxf(rm, __shfl_xor(rm, mm, 32));
      float mn = fmaxf(mrow[r], rm);
      float corr = __expf(mrow[r] - mn);
      mrow[r] = mn;
      float e0 = __expf(s0 - mn), e1 = __expf(s1 - mn);
      float rs = e0 + e1;
#pragma unroll
      for (int mm = 1; mm < 16; mm <<= 1) rs += __shfl_xor(rs, mm, 32);
      lrow[r] = lrow[r] * corr + rs;
      p0[r] = e0; p1[r] = e1;
#pragma unroll
      for (int nc = 0; nc < 4; ++nc) o[nc][r] *= corr;
    }
    // stage P (16x32 bf16) row-major in LDS, then reload as A fragment
#pragma unroll
    for (int r = 0; r < 8; ++r) {
      P[(mb + r) * 32 + n] = (bf16)p0[r];
      P[(mb + r) * 32 + 16 + n] = (bf16)p1[r];
    }
    asm volatile("s_wait_dscnt 0" ::: "memory");  // same-wave LDS RAW
    v16bf pa = load_a(P, 32, 0, 0);
#pragma unroll
    for (int nc = 0; nc < 4; ++nc)
      o[nc] = wmma_bf16(pa, load_b(vp, SEQ, nc * 16, key0), o[nc]);
  }

#pragma unroll
  for (int nc = 0; nc < 4; ++nc)
#pragma unroll
    for (int r = 0; r < 8; ++r)
      head[((size_t)b * SEQ + m0 + mb + r) * DK + nc * 16 + n] =
          (bf16)(o[nc][r] / lrow[r]);
}

// ---------------------------------------------------------------------------
// mha = head @ WoSum + bo + residual(x)  ->  pre (fp32, pre-LayerNorm)
// ---------------------------------------------------------------------------
__global__ __launch_bounds__(256) void proj_kernel(const bf16* head,
                                                   const bf16* woST,
                                                   const float* bo,
                                                   const float* x, float* pre) {
  int wave = threadIdx.x >> 5;
  int gw = blockIdx.x * 8 + wave;   // 0..8191
  int m0 = (gw >> 3) * 16, ng = gw & 7;
  v16bf a0 = load_a(head, DK, m0, 0);
  v16bf a1 = load_a(head, DK, m0, 32);
  int l = threadIdx.x & 31, n = l & 15, mb = (l >> 4) * 8;
#pragma unroll
  for (int nt = 0; nt < 4; ++nt) {
    int n0 = ng * 64 + nt * 16;
    v8f acc = (v8f)0.f;
    acc = wmma_bf16(a0, load_b(woST, DK, n0, 0), acc);
    acc = wmma_bf16(a1, load_b(woST, DK, n0, 32), acc);
#pragma unroll
    for (int r = 0; r < 8; ++r) {
      size_t row = m0 + mb + r; int col = n0 + n;
      pre[row * DMODEL + col] = acc[r] + bo[col] + x[row * DMODEL + col];
    }
  }
}

// ---------------------------------------------------------------------------
// Whole-tensor LayerNorm (deterministic 2-level tree reduce, no atomics)
// ---------------------------------------------------------------------------
__global__ __launch_bounds__(256) void ln_reduce(const float* src, float* part) {
  __shared__ float ss[256], sq[256];
  float s = 0.f, q = 0.f;
  for (long i = (long)blockIdx.x * 256 + threadIdx.x; i < NELEM;
       i += (long)gridDim.x * 256) {
    float v = src[i]; s += v; q += v * v;
  }
  ss[threadIdx.x] = s; sq[threadIdx.x] = q;
  __syncthreads();
  for (int st = 128; st > 0; st >>= 1) {
    if ((int)threadIdx.x < st) { ss[threadIdx.x] += ss[threadIdx.x + st]; sq[threadIdx.x] += sq[threadIdx.x + st]; }
    __syncthreads();
  }
  if (threadIdx.x == 0) { part[blockIdx.x] = ss[0]; part[gridDim.x + blockIdx.x] = sq[0]; }
}

__global__ __launch_bounds__(256) void ln_final(const float* part, float* stats) {
  __shared__ float ss[256], sq[256];
  int t = threadIdx.x;
  ss[t] = part[t]; sq[t] = part[256 + t];
  __syncthreads();
  for (int st = 128; st > 0; st >>= 1) {
    if (t < st) { ss[t] += ss[t + st]; sq[t] += sq[t + st]; }
    __syncthreads();
  }
  if (t == 0) {
    float cnt = (float)NELEM;
    float mean = ss[0] / cnt;
    float var = sq[0] / cnt - mean * mean;
    stats[0] = mean;
    stats[1] = rsqrtf(var + LN_EPS);
  }
}

__global__ __launch_bounds__(256) void ln_apply(const float* pre,
                                                const float* stats, float* xo,
                                                bf16* xb) {
  float mean = stats[0], rstd = stats[1];
  for (long i = (long)blockIdx.x * blockDim.x + threadIdx.x; i < NELEM;
       i += (long)gridDim.x * blockDim.x) {
    float v = (pre[i] - mean) * rstd;
    xo[i] = v;
    xb[i] = (bf16)v;
  }
}

// ---------------------------------------------------------------------------
// Fused FF: per 16-row tile (1 block, 8 waves).
//  - x1 A-tile (16x512 bf16 = 16KB) staged ONCE into LDS with
//    GLOBAL_LOAD_ASYNC_TO_LDS_B128 (ASYNCcnt), shared by all 8 waves.
//  - hidden = relu(x1@W1+b1) staged as bf16 in 64KB LDS
//  - out = hidden@W2 + b2 + residual(x1) -> pre2
// ---------------------------------------------------------------------------
__global__ __launch_bounds__(256) void ff_kernel(const bf16* x1b, const float* x1,
                                                 const bf16* w1T, const float* b1,
                                                 const bf16* w2T, const float* b2,
                                                 float* pre2) {
  __shared__ __align__(16) bf16 xtile[16 * DMODEL]; // 16 KB
  __shared__ __align__(16) bf16 hid[16 * DFF];      // 64 KB
  int wave = threadIdx.x >> 5;
  int m0 = blockIdx.x * 16;
  int l = threadIdx.x & 31, n = l & 15, mb = (l >> 4) * 8;

  // cooperative async stage of the shared A-tile: 16KB = 1024 x 16B chunks
  {
    const char* src = (const char*)(x1b + (size_t)m0 * DMODEL);
#pragma unroll
    for (int c = 0; c < 4; ++c) {
      int chunk = threadIdx.x + c * 256;               // 0..1023
      unsigned long long ga =
          (unsigned long long)(uintptr_t)(src + (size_t)chunk * 16);
      unsigned int la =
          (unsigned int)(uintptr_t)((char*)xtile + (size_t)chunk * 16);
      asm volatile("global_load_async_to_lds_b128 %0, %1, off"
                   :: "v"(la), "v"(ga) : "memory");
    }
    // prefetch this wave's phase-2 W2^T rows into cache while we compute
    __builtin_prefetch(w2T + ((size_t)wave * 64 + (threadIdx.x & 31)) * DFF, 0, 1);
    asm volatile("s_wait_asynccnt 0" ::: "memory");
  }
  __syncthreads();

  // phase 1: hidden tiles (128 n-tiles of 16 / 8 waves), A from LDS
  for (int t = 0; t < 16; ++t) {
    int n0 = (wave * 16 + t) * 16;
    v8f acc = (v8f)0.f;
    for (int kk = 0; kk < DMODEL; kk += 32)
      acc = wmma_bf16(load_a(xtile, DMODEL, 0, kk), load_b(w1T, DMODEL, n0, kk), acc);
#pragma unroll
    for (int r = 0; r < 8; ++r)
      hid[(mb + r) * DFF + n0 + n] = (bf16)fmaxf(acc[r] + b1[n0 + n], 0.0f);
  }
  __syncthreads();
  // phase 2: output tiles (32 n-tiles of 16 / 8 waves), A from LDS
#pragma unroll
  for (int t = 0; t < 4; ++t) {
    int n0 = (wave * 4 + t) * 16;
    v8f acc = (v8f)0.f;
    for (int kk = 0; kk < DFF; kk += 32)
      acc = wmma_bf16(load_a(hid, DFF, 0, kk), load_b(w2T, DFF, n0, kk), acc);
#pragma unroll
    for (int r = 0; r < 8; ++r) {
      size_t row = m0 + mb + r; int col = n0 + n;
      pre2[row * DMODEL + col] = acc[r] + b2[col] + x1[row * DMODEL + col];
    }
  }
}

// ---------------------------------------------------------------------------
// Host launch
// ---------------------------------------------------------------------------
extern "C" void kernel_launch(void* const* d_in, const int* in_sizes, int n_in,
                              void* d_out, int out_size, void* d_ws, size_t ws_size,
                              hipStream_t stream) {
  const int*   ids = (const int*)d_in[0];
  const float* emb = (const float*)d_in[1];
  const float* Wq = (const float*)d_in[2];  const float* bq = (const float*)d_in[3];
  const float* Wk = (const float*)d_in[4];  const float* bk = (const float*)d_in[5];
  const float* Wv = (const float*)d_in[6];  const float* bv = (const float*)d_in[7];
  const float* Wo = (const float*)d_in[8];  const float* bo = (const float*)d_in[9];
  const float* W1 = (const float*)d_in[10]; const float* b1 = (const float*)d_in[11];
  const float* W2 = (const float*)d_in[12]; const float* b2 = (const float*)d_in[13];

  char* w = (char*)d_ws;
  auto carve = [&](size_t bytes) {
    void* p = (void*)w;
    w += (bytes + 255) & ~(size_t)255;
    return p;
  };
  float* x    = (float*)carve((size_t)NELEM * 4);
  float* pre  = (float*)carve((size_t)NELEM * 4);
  float* x1   = (float*)carve((size_t)NELEM * 4);
  bf16*  xb   = (bf16*) carve((size_t)NELEM * 2);
  bf16*  x1b  = (bf16*) carve((size_t)NELEM * 2);
  bf16*  qbuf = (bf16*) carve((size_t)MTOT * DK * 2);
  bf16*  kbuf = (bf16*) carve((size_t)MTOT * DK * 2);
  bf16*  vT   = (bf16*) carve((size_t)MTOT * DK * 2);
  bf16*  head = (bf16*) carve((size_t)MTOT * DK * 2);
  bf16*  wqT  = (bf16*) carve(64L * 512 * 2);
  bf16*  wkT  = (bf16*) carve(64L * 512 * 2);
  bf16*  wvT  = (bf16*) carve(64L * 512 * 2);
  bf16*  woST = (bf16*) carve(512L * 64 * 2);
  bf16*  w1T  = (bf16*) carve(2048L * 512 * 2);
  bf16*  w2T  = (bf16*) carve(512L * 2048 * 2);
  float* part = (float*)carve(512 * 4);
  float* stats= (float*)carve(2 * 4);

  prep_weights<<<128, 256, 0, stream>>>(Wq, Wk, Wv, Wo, W1, W2,
                                        wqT, wkT, wvT, woST, w1T, w2T);
  embed_pe<<<1024, 256, 0, stream>>>(ids, emb, x, xb);

  for (int rep = 0; rep < 6; ++rep) {
    qkv_kernel<<<256, 128, 0, stream>>>(xb, wqT, wkT, wvT, bq, bk, bv,
                                        qbuf, kbuf, vT);
    attn_kernel<<<256, 128, 0, stream>>>(qbuf, kbuf, vT, head);
    proj_kernel<<<1024, 256, 0, stream>>>(head, woST, bo, x, pre);
    ln_reduce<<<256, 256, 0, stream>>>(pre, part);
    ln_final<<<1, 256, 0, stream>>>(part, stats);
    ln_apply<<<1024, 256, 0, stream>>>(pre, stats, x1, x1b);
    ff_kernel<<<1024, 256, 0, stream>>>(x1b, x1, w1T, b1, w2T, b2, pre);
    ln_reduce<<<256, 256, 0, stream>>>(pre, part);
    ln_final<<<1, 256, 0, stream>>>(part, stats);
    float* xo = (rep == 5) ? (float*)d_out : x;
    ln_apply<<<1024, 256, 0, stream>>>(pre, stats, xo, xb);
  }
}
